// WireframeGNN_10943576671010
// MI455X (gfx1250) — matrix operand order repaired and checked by reading, not compiled, not run on hardware
//
#include <hip/hip_runtime.h>

// ---------------------------------------------------------------------------
// WireframeGNN on MI455X (gfx1250): 2x GCNConv+BN+ReLU + FC.
// GEMMs: bf16 WMMA (v_wmma_f32_16x16x32_bf16), 128x64 block tile,
//        32x32 C per wave (4 WMMA / K-step), float4 global staging.
// Edge aggregation: L2-resident global_atomic_add_f32 scatter.
// ---------------------------------------------------------------------------

typedef __attribute__((ext_vector_type(16))) __bf16 v16bf;
typedef __attribute__((ext_vector_type(8)))  __bf16 v8bf;
typedef __attribute__((ext_vector_type(4)))  __bf16 v4bf;
typedef __attribute__((ext_vector_type(8)))  float  v8f;

#define LPAD 40   // row stride in halves: 80 B = 16B-aligned, bank-conflict free

__device__ __forceinline__ void atomic_add_f32(float* p, float v) {
  __hip_atomic_fetch_add(p, v, __ATOMIC_RELAXED, __HIP_MEMORY_SCOPE_AGENT);
}

__device__ __forceinline__ v16bf load_fragA(const __bf16* p) {
  // A 16x32 bf16 frag: K = 8*lanehalf + (e&7) + 16*(e>>3)  (ISA 7.12.2)
  union { v16bf v; v8bf h[2]; } u;
  u.h[0] = *(const v8bf*)p;
  u.h[1] = *(const v8bf*)(p + 16);
  return u.v;
}
__device__ __forceinline__ v16bf load_fragB(const __bf16* p) {
  // B 32x16 bf16 frag (LDS holds B transposed [n][k]): K = 8*lanehalf + (e&7) + 16*(e>>3)
  union { v16bf v; v8bf h[2]; } u;
  u.h[0] = *(const v8bf*)p;
  u.h[1] = *(const v8bf*)(p + 16);
  return u.v;
}

// ------------------------- WMMA GEMM --------------------------------------
// out[M,Ntot] = relu?( concat(A0|A1|A2)[M,K] @ W[K,Ntot] + bias )
// K0/K1/K2 are multiples of 32 -> every 32-wide K tile lies in one segment,
// so the segment select is a uniform scalar branch per K-step.
__global__ __launch_bounds__(256) void gemm_wmma_bf16(
    const float* __restrict__ A0, const float* __restrict__ A1,
    const float* __restrict__ A2, int K0, int K1, int K2,
    const float* __restrict__ W, const float* __restrict__ bias,
    float* __restrict__ out, int M, int Ntot, int relu) {
  __shared__ __align__(16) __bf16 ldsA[128 * LPAD];   // [row][k]
  __shared__ __align__(16) __bf16 ldsB[64 * LPAD];    // transposed [n][k]

  const int K     = K0 + K1 + K2;
  const int tid   = threadIdx.x;
  const int lane  = tid & 31;
  const int wave  = tid >> 5;
  const int wm    = wave & 3;           // M sub-tile (32 rows each)
  const int wn    = wave >> 2;          // N sub-tile (32 cols each)
  const int lhalf = lane >> 4;
  const int l15   = lane & 15;
  const int rowBase = blockIdx.x * 128;
  const int colBase = blockIdx.y * 64;
  const bool full = (rowBase + 128 <= M);

  v8f c00 = {}, c01 = {}, c10 = {}, c11 = {};

  for (int kb = 0; kb < K; kb += 32) {
    // ---- uniform (scalar) segment select for the virtual concat
    const float* __restrict__ Aseg;
    int segK, kseg;
    if (kb < K0)           { Aseg = A0; segK = K0; kseg = kb; }
    else if (kb < K0 + K1) { Aseg = A1; segK = K1; kseg = kb - K0; }
    else                   { Aseg = A2; segK = K2; kseg = kb - K0 - K1; }

    // ---- stage A 128x32: float4 loads -> packed bf16 -> ds_store_b64
    if (full) {
#pragma unroll
      for (int i = 0; i < 4; ++i) {
        int q = tid + i * 256;            // 1024 quads
        int r = q >> 3, kq = (q & 7) * 4;
        float4 f = *(const float4*)&Aseg[(size_t)(rowBase + r) * segK + kseg + kq];
        v4bf h = { (__bf16)f.x, (__bf16)f.y, (__bf16)f.z, (__bf16)f.w };
        *(v4bf*)&ldsA[r * LPAD + kq] = h;
      }
    } else {
#pragma unroll
      for (int i = 0; i < 4; ++i) {
        int q = tid + i * 256;
        int r = q >> 3, kq = (q & 7) * 4;
        int row = rowBase + r;
        float4 f = make_float4(0.f, 0.f, 0.f, 0.f);
        if (row < M)
          f = *(const float4*)&Aseg[(size_t)row * segK + kseg + kq];
        v4bf h = { (__bf16)f.x, (__bf16)f.y, (__bf16)f.z, (__bf16)f.w };
        *(v4bf*)&ldsA[r * LPAD + kq] = h;
      }
    }

    // ---- stage B 32x64 transposed: float4 along n, 4 scalar b16 stores
#pragma unroll
    for (int i = 0; i < 2; ++i) {
      int q = tid + i * 256;              // 512 quads
      int k = q >> 4, nq = (q & 15) * 4;
      float4 f = *(const float4*)&W[(size_t)(kb + k) * Ntot + colBase + nq];
      ldsB[(nq + 0) * LPAD + k] = (__bf16)f.x;
      ldsB[(nq + 1) * LPAD + k] = (__bf16)f.y;
      ldsB[(nq + 2) * LPAD + k] = (__bf16)f.z;
      ldsB[(nq + 3) * LPAD + k] = (__bf16)f.w;
    }
    if (kb + 32 < K && lane == 0)         // warm next W tile
      __builtin_prefetch(&W[(size_t)(kb + 32) * Ntot + colBase], 0, 1);
    __syncthreads();

    // ---- 4 WMMAs per wave per K-step (2 A frags x 2 B frags)
    const __bf16* a0p = &ldsA[(wm * 32 + l15) * LPAD + lhalf * 8];
    const __bf16* b0p = &ldsB[(wn * 32 + l15) * LPAD + lhalf * 8];
    v16bf a0 = load_fragA(a0p);
    v16bf a1 = load_fragA(a0p + 16 * LPAD);
    v16bf b0 = load_fragB(b0p);
    v16bf b1 = load_fragB(b0p + 16 * LPAD);

    c00 = __builtin_amdgcn_wmma_f32_16x16x32_bf16(false, a0, false, b0, (short)0, c00, false, false);
    c01 = __builtin_amdgcn_wmma_f32_16x16x32_bf16(false, a0, false, b1, (short)0, c01, false, false);
    c10 = __builtin_amdgcn_wmma_f32_16x16x32_bf16(false, a1, false, b0, (short)0, c10, false, false);
    c11 = __builtin_amdgcn_wmma_f32_16x16x32_bf16(false, a1, false, b1, (short)0, c11, false, false);
    __syncthreads();
  }

  // ---- epilogue: C/D layout VGPR v -> M = v + 8*lhalf, N = l15
#pragma unroll
  for (int i = 0; i < 2; ++i) {
#pragma unroll
    for (int j = 0; j < 2; ++j) {
      v8f& c = (i == 0) ? (j == 0 ? c00 : c01) : (j == 0 ? c10 : c11);
      int col = colBase + wn * 32 + j * 16 + l15;
      float bi = bias ? bias[col] : 0.f;
#pragma unroll
      for (int v = 0; v < 8; ++v) {
        int row = rowBase + wm * 32 + i * 16 + lhalf * 8 + v;
        if (full || row < M) {
          float r = c[v] + bi;
          if (relu) r = fmaxf(r, 0.f);
          out[(size_t)row * Ntot + col] = r;
        }
      }
    }
  }
}

// ------------------------- small helpers ----------------------------------
__global__ void fill_f32(float* p, float v, int n) {
  int i = blockIdx.x * blockDim.x + threadIdx.x;
  if (i < n) p[i] = v;
}

__global__ void deg_accum(const int* __restrict__ dst, int E, float* deg) {
  int e = blockIdx.x * blockDim.x + threadIdx.x;
  if (e < E) atomic_add_f32(&deg[dst[e]], 1.0f);
}

__global__ void rsqrt_ip(float* p, int n) {
  int i = blockIdx.x * blockDim.x + threadIdx.x;
  if (i < n) p[i] = rsqrtf(p[i]);
}

// agg[i,f] = bias[f] + h[i,f] * dinv[i]^2   (self-loop term + bias)
__global__ void agg_init(const float* __restrict__ h, const float* __restrict__ dinv,
                         const float* __restrict__ bias, float* __restrict__ agg,
                         int n, int F) {
  int i = blockIdx.x * blockDim.x + threadIdx.x;
  if (i < n * F) {
    int node = i / F, f = i - node * F;
    float di = dinv[node];
    agg[i] = bias[f] + h[i] * di * di;
  }
}

// one block per edge, one lane per feature; scatter-add stays L2-resident
__global__ void edge_agg(const int* __restrict__ src, const int* __restrict__ dst,
                         const float* __restrict__ h, const float* __restrict__ dinv,
                         float* __restrict__ agg, int F) {
  int e = blockIdx.x;
  int f = threadIdx.x;
  int s = src[e], d = dst[e];
  float norm = dinv[s] * dinv[d];
  atomic_add_f32(&agg[(size_t)d * F + f], h[(size_t)s * F + f] * norm);
}

// per-block register accumulation over 512 rows, then one atomic per column
__global__ void col_stats(const float* __restrict__ x, int n, int F,
                          float* __restrict__ sum, float* __restrict__ sumsq) {
  int f = threadIdx.x;
  int r0 = blockIdx.x * 512;
  int r1 = r0 + 512; if (r1 > n) r1 = n;
  float s = 0.f, q = 0.f;
  for (int r = r0; r < r1; ++r) {
    float v = x[(size_t)r * F + f];
    s += v; q += v * v;
  }
  atomic_add_f32(&sum[f], s);
  atomic_add_f32(&sumsq[f], q);
}

__global__ void bn_relu(const float* __restrict__ x, const float* __restrict__ sum,
                        const float* __restrict__ sumsq, const float* __restrict__ g,
                        const float* __restrict__ be, float* __restrict__ out,
                        int n, int F) {
  int i = blockIdx.x * blockDim.x + threadIdx.x;
  if (i < n * F) {
    int f = i % F;
    float inv_n = 1.0f / (float)n;
    float mean  = sum[f] * inv_n;
    float var   = sumsq[f] * inv_n - mean * mean;
    float r = (x[i] - mean) * rsqrtf(var + 1e-5f) * g[f] + be[f];
    out[i] = fmaxf(r, 0.f);
  }
}

// ------------------------- orchestration ----------------------------------
extern "C" void kernel_launch(void* const* d_in, const int* in_sizes, int n_in,
                              void* d_out, int out_size, void* d_ws, size_t ws_size,
                              hipStream_t stream) {
  const float* X   = (const float*)d_in[0];
  const int*   edg = (const int*)d_in[1];
  const float* W1  = (const float*)d_in[2];
  const float* b1  = (const float*)d_in[3];
  const float* g1  = (const float*)d_in[4];
  const float* be1 = (const float*)d_in[5];
  const float* W2  = (const float*)d_in[6];
  const float* b2  = (const float*)d_in[7];
  const float* g2  = (const float*)d_in[8];
  const float* be2 = (const float*)d_in[9];
  const float* Wfc = (const float*)d_in[10];
  const float* bfc = (const float*)d_in[11];
  float* out = (float*)d_out;

  const int D0 = 256, D1 = 128, D2 = 64;
  const int N = in_sizes[0] / D0;          // 100000
  const int E = in_sizes[1] / 2;           // 1600000
  const int* src = edg;
  const int* dst = edg + E;

  // workspace carve-up (floats)
  float* ws    = (float*)d_ws;
  float* dinv  = ws;                        // N
  float* bufA  = dinv + N;                  // N*128 : h1, then x1
  float* bufB  = bufA + (size_t)N * D1;     // N*128 : agg1
  float* bufC  = bufB + (size_t)N * D1;     // N*64  : h2, then x2
  float* bufD  = bufC + (size_t)N * D2;     // N*64  : agg2
  float* stats = bufD + (size_t)N * D2;     // 2*128 + 2*64

  // ---- symmetric normalization: deg (with self loop) -> 1/sqrt(deg)
  fill_f32<<<(N + 255) / 256, 256, 0, stream>>>(dinv, 1.0f, N);
  deg_accum<<<(E + 255) / 256, 256, 0, stream>>>(dst, E, dinv);
  rsqrt_ip<<<(N + 255) / 256, 256, 0, stream>>>(dinv, N);

  // ---- layer 1: h1 = X @ W1 ; aggregate ; BN+ReLU -> x1 (bufA)
  gemm_wmma_bf16<<<dim3((N + 127) / 128, D1 / 64), 256, 0, stream>>>(
      X, nullptr, nullptr, D0, 0, 0, W1, nullptr, bufA, N, D1, 0);
  agg_init<<<((N * D1) + 255) / 256, 256, 0, stream>>>(bufA, dinv, b1, bufB, N, D1);
  edge_agg<<<E, D1, 0, stream>>>(src, dst, bufA, dinv, bufB, D1);
  fill_f32<<<1, 256, 0, stream>>>(stats, 0.f, 2 * D1);
  col_stats<<<(N + 511) / 512, D1, 0, stream>>>(bufB, N, D1, stats, stats + D1);
  bn_relu<<<((N * D1) + 255) / 256, 256, 0, stream>>>(
      bufB, stats, stats + D1, g1, be1, bufA, N, D1);

  // ---- layer 2: h2 = x1 @ W2 ; aggregate ; BN+ReLU -> x2 (bufC)
  gemm_wmma_bf16<<<dim3((N + 127) / 128, D2 / 64), 256, 0, stream>>>(
      bufA, nullptr, nullptr, D1, 0, 0, W2, nullptr, bufC, N, D2, 0);
  agg_init<<<((N * D2) + 255) / 256, 256, 0, stream>>>(bufC, dinv, b2, bufD, N, D2);
  edge_agg<<<E, D2, 0, stream>>>(src, dst, bufC, dinv, bufD, D2);
  fill_f32<<<1, 128, 0, stream>>>(stats + 2 * D1, 0.f, 2 * D2);
  col_stats<<<(N + 511) / 512, D2, 0, stream>>>(bufD, N, D2,
                                               stats + 2 * D1, stats + 2 * D1 + D2);
  bn_relu<<<((N * D2) + 255) / 256, 256, 0, stream>>>(
      bufD, stats + 2 * D1, stats + 2 * D1 + D2, g2, be2, bufC, N, D2);

  // ---- FC over virtual concat [X | x1 | x2] : out = relu(cat @ Wfc + bfc)
  gemm_wmma_bf16<<<dim3((N + 127) / 128, D0 / 64), 256, 0, stream>>>(
      X, bufA, bufC, D0, D1, D2, Wfc, bfc, out, N, D0, 1);
}
